// CasualSelfAttention_4595615007185
// MI455X (gfx1250) — matrix operand order repaired and checked by reading, not compiled, not run on hardware
//
#include <hip/hip_runtime.h>
#include <hip/hip_bf16.h>

// ---------------------------------------------------------------------------
// Causal self-attention for B=2, T=4096, C=768, H=12, D=64 on gfx1250 (CDNA5)
// All matmuls via v_wmma_f32_16x16x32_bf16, fp32 accumulate.
// Explicit ping-pong double buffering (2x-unrolled K loops, no rotation movs).
// ---------------------------------------------------------------------------

typedef __attribute__((ext_vector_type(16))) __bf16 v16bf;
typedef __attribute__((ext_vector_type(8)))  __bf16 v8bf;
typedef __attribute__((ext_vector_type(8)))  float  v8f;

#define B_   2
#define T_   4096
#define C_   768
#define H_   12
#define D_   64
#define C3_  2304   // 3*C

static __device__ __forceinline__ v8f wmma_bf16(v16bf a, v16bf b, v8f c) {
  // D = A(16x32 bf16) * B(32x16 bf16) + C(16x16 f32)
  return __builtin_amdgcn_wmma_f32_16x16x32_bf16(
      /*neg_a=*/false, a, /*neg_b=*/false, b,
      /*c_mod=*/(short)0, c, /*reuse_a=*/false, /*reuse_b=*/false);
}

// A-fragment (16x32, M x K): lane holds row M = lane%16.
// vec[0..7]  = K = 16*hi + (0..7)
// vec[8..15] = K = 16*hi + 8 + (0..7)
static __device__ __forceinline__ v16bf load_a_frag(const __bf16* row_k0, int hi) {
  v8bf c0 = *(const v8bf*)(row_k0 + 16 * hi);
  v8bf c1 = *(const v8bf*)(row_k0 + 16 * hi + 8);
  return __builtin_shufflevector(c0, c1, 0,1,2,3,4,5,6,7,8,9,10,11,12,13,14,15);
}

// ---------------------------------------------------------------------------
// Converters
// ---------------------------------------------------------------------------
__global__ __launch_bounds__(256) void k_f32_to_bf16(const float* __restrict__ in,
                                                     __bf16* __restrict__ out, int n) {
  int i = blockIdx.x * 256 + threadIdx.x;
  if (i < n) out[i] = (__bf16)in[i];
}

// in: [rows_in][cols_in] f32  ->  out: [cols_in][rows_in] bf16
__global__ __launch_bounds__(256) void k_transpose_bf16(const float* __restrict__ in,
                                                        __bf16* __restrict__ out,
                                                        int rows_in, int cols_in) {
  int i = blockIdx.x * 256 + threadIdx.x;
  if (i < rows_in * cols_in) {
    int c = i / rows_in;
    int r = i - c * rows_in;
    out[i] = (__bf16)in[r * cols_in + c];
  }
}

// ---------------------------------------------------------------------------
// Shared GEMM core: 32x64 tile per wave, ping-pong double buffering.
// A: [*, C_] row-major bf16; B: wT [N, C_] row-major bf16 (pre-transposed).
// ---------------------------------------------------------------------------
static __device__ __forceinline__ void gemm_32x64_core(
    const __bf16* __restrict__ arow0, const __bf16* __restrict__ arow1,
    const __bf16* const brow[4], int hi, v8f acc[8]) {
  // prologue: buffer A <- k=0
  v16bf aA0 = load_a_frag(arow0, hi);
  v16bf aA1 = load_a_frag(arow1, hi);
  v16bf bA0 = *(const v16bf*)(brow[0]);
  v16bf bA1 = *(const v16bf*)(brow[1]);
  v16bf bA2 = *(const v16bf*)(brow[2]);
  v16bf bA3 = *(const v16bf*)(brow[3]);

  for (int k0 = 0; k0 < C_; k0 += 64) {
    // prefetch buffer B <- k0+32 (always valid: C_ % 64 == 0)
    v16bf aB0 = load_a_frag(arow0 + k0 + 32, hi);
    v16bf aB1 = load_a_frag(arow1 + k0 + 32, hi);
    v16bf bB0 = *(const v16bf*)(brow[0] + k0 + 32);
    v16bf bB1 = *(const v16bf*)(brow[1] + k0 + 32);
    v16bf bB2 = *(const v16bf*)(brow[2] + k0 + 32);
    v16bf bB3 = *(const v16bf*)(brow[3] + k0 + 32);

    // compute k0 with buffer A
    acc[0] = wmma_bf16(aA0, bA0, acc[0]);  acc[4] = wmma_bf16(aA1, bA0, acc[4]);
    acc[1] = wmma_bf16(aA0, bA1, acc[1]);  acc[5] = wmma_bf16(aA1, bA1, acc[5]);
    acc[2] = wmma_bf16(aA0, bA2, acc[2]);  acc[6] = wmma_bf16(aA1, bA2, acc[6]);
    acc[3] = wmma_bf16(aA0, bA3, acc[3]);  acc[7] = wmma_bf16(aA1, bA3, acc[7]);

    // prefetch buffer A <- k0+64 (clamped: always in-bounds)
    const int k2 = (k0 + 64 < C_) ? (k0 + 64) : 0;
    aA0 = load_a_frag(arow0 + k2, hi);
    aA1 = load_a_frag(arow1 + k2, hi);
    bA0 = *(const v16bf*)(brow[0] + k2);
    bA1 = *(const v16bf*)(brow[1] + k2);
    bA2 = *(const v16bf*)(brow[2] + k2);
    bA3 = *(const v16bf*)(brow[3] + k2);

    // compute k0+32 with buffer B
    acc[0] = wmma_bf16(aB0, bB0, acc[0]);  acc[4] = wmma_bf16(aB1, bB0, acc[4]);
    acc[1] = wmma_bf16(aB0, bB1, acc[1]);  acc[5] = wmma_bf16(aB1, bB1, acc[5]);
    acc[2] = wmma_bf16(aB0, bB2, acc[2]);  acc[6] = wmma_bf16(aB1, bB2, acc[6]);
    acc[3] = wmma_bf16(aB0, bB3, acc[3]);  acc[7] = wmma_bf16(aB1, bB3, acc[7]);
  }
}

// ---------------------------------------------------------------------------
// QKV GEMM: [B*T, C] x [C, 3C] (+bias), scatter into Q[B,H,T,D], K[B,H,T,D],
// Vt[B,H,D,T]. wT is W_attn^T [3C][C].
// ---------------------------------------------------------------------------
__global__ __launch_bounds__(256) void k_qkv_gemm(
    const __bf16* __restrict__ xbf, const __bf16* __restrict__ wT,
    const float* __restrict__ bias,
    __bf16* __restrict__ Qo, __bf16* __restrict__ Ko, __bf16* __restrict__ Vt) {
  const int tiles_n = C3_ / 64;                 // 36
  const int tm = blockIdx.x / tiles_n;
  const int tn = blockIdx.x % tiles_n;
  const int w    = threadIdx.x >> 5;
  const int lane = threadIdx.x & 31;
  const int col  = lane & 15;
  const int hi   = lane >> 4;
  const int mbase = tm * 256 + w * 32;
  const int nbase = tn * 64;

  const __bf16* arow0 = xbf + (size_t)(mbase + col) * C_;
  const __bf16* arow1 = xbf + (size_t)(mbase + 16 + col) * C_;
  const __bf16* brow[4];
#pragma unroll
  for (int nt = 0; nt < 4; ++nt)
    brow[nt] = wT + (size_t)(nbase + nt * 16 + col) * C_ + 16 * hi;

  v8f acc[8] = {};
  gemm_32x64_core(arow0, arow1, brow, hi, acc);

#pragma unroll
  for (int g = 0; g < 2; ++g) {
#pragma unroll
    for (int nt = 0; nt < 4; ++nt) {
      const int n  = nbase + nt * 16 + col;     // 0..3C-1
      const float bv = bias[n];
      const int seg = n / C_;                   // uniform over the 16-col tile
      const int nn  = n - seg * C_;
      const int h   = nn >> 6;
      const int d   = nn & 63;
#pragma unroll
      for (int r = 0; r < 8; ++r) {
        const int bt = mbase + g * 16 + r + 8 * hi;
        const int b  = bt >> 12;                // /T_
        const int t  = bt & (T_ - 1);
        const __bf16 v = (__bf16)(acc[g * 4 + nt][r] + bv);
        if (seg == 0)      Qo[((size_t)(b * H_ + h) * T_ + t) * D_ + d] = v;
        else if (seg == 1) Ko[((size_t)(b * H_ + h) * T_ + t) * D_ + d] = v;
        else               Vt[((size_t)(b * H_ + h) * D_ + d) * T_ + t] = v;
      }
    }
  }
}

// ---------------------------------------------------------------------------
// Flash-style causal attention. Block = 8 waves = 128 query rows of one
// (b, h). Each wave owns 16 rows; streams 32 keys/step with its own causal
// trip count (no block barriers; LDS region is per-wave private).
// Pair-loop ping-pong on K fragments; V frags issued before the softmax
// VALU section so their latency is hidden.
// ---------------------------------------------------------------------------
__global__ __launch_bounds__(256) void k_attn(
    const __bf16* __restrict__ Q, const __bf16* __restrict__ K,
    const __bf16* __restrict__ Vt, __bf16* __restrict__ Y) {
  __shared__ __bf16 plds[8][16][32];            // per-wave 16x32 P staging (8 KB)

  const int qblocks = T_ / 128;                 // 32
  const int qblk = blockIdx.x % qblocks;
  const int h    = (blockIdx.x / qblocks) % H_;
  const int b    = blockIdx.x / (qblocks * H_);
  const int w    = threadIdx.x >> 5;
  const int lane = threadIdx.x & 31;
  const int col  = lane & 15;
  const int hi   = lane >> 4;
  const int qbase = qblk * 128 + w * 16;

  const size_t head = (size_t)(b * H_ + h) * T_ * D_;
  const __bf16* Qh = Q + head;
  const __bf16* Kh = K + head;
  const __bf16* Vh = Vt + head;                 // [D][T] per head

  // Q fragments for this wave's 16 rows (d 0..31 and 32..63)
  const __bf16* qrow = Qh + (size_t)(qbase + col) * D_;
  const v16bf qa0 = load_a_frag(qrow, hi);
  const v16bf qa1 = load_a_frag(qrow + 32, hi);

  float m[8], l[8];
  v8f O[4] = {};
#pragma unroll
  for (int r = 0; r < 8; ++r) { m[r] = -1e30f; l[r] = 0.0f; }

  const int kend = qbase + 16;                  // keys needed: [0, qbase+15]

  // one 32-key step: S = Q K^T, streaming softmax, O += P V
  auto step = [&](int kbase, v16bf kb0, v16bf kb1, v16bf kb2, v16bf kb3) {
    // V fragments for this step (consumed after softmax)
    v16bf vb0 = *(const v16bf*)(Vh + (size_t)(0 * 16 + col) * T_ + kbase + 16 * hi);
    v16bf vb1 = *(const v16bf*)(Vh + (size_t)(1 * 16 + col) * T_ + kbase + 16 * hi);
    v16bf vb2 = *(const v16bf*)(Vh + (size_t)(2 * 16 + col) * T_ + kbase + 16 * hi);
    v16bf vb3 = *(const v16bf*)(Vh + (size_t)(3 * 16 + col) * T_ + kbase + 16 * hi);

    v8f s0 = {}, s1 = {};
    s0 = wmma_bf16(qa0, kb0, s0);
    s0 = wmma_bf16(qa1, kb1, s0);
    s1 = wmma_bf16(qa0, kb2, s1);
    s1 = wmma_bf16(qa1, kb3, s1);

    const int k0c = kbase + col;
    const int k1c = kbase + 16 + col;
#pragma unroll
    for (int r = 0; r < 8; ++r) {
      const int qi = qbase + r + 8 * hi;
      float v0 = (k0c <= qi) ? s0[r] * 0.125f : -1e30f;   // 1/sqrt(64)
      float v1 = (k1c <= qi) ? s1[r] * 0.125f : -1e30f;
      float tmax = fmaxf(v0, v1);
      tmax = fmaxf(tmax, __shfl_xor(tmax, 1));
      tmax = fmaxf(tmax, __shfl_xor(tmax, 2));
      tmax = fmaxf(tmax, __shfl_xor(tmax, 4));
      tmax = fmaxf(tmax, __shfl_xor(tmax, 8));
      const float mn = fmaxf(m[r], tmax);
      const float corr = __expf(m[r] - mn);
      m[r] = mn;
      const float p0 = __expf(v0 - mn);
      const float p1 = __expf(v1 - mn);
      float ps = p0 + p1;
      ps += __shfl_xor(ps, 1);
      ps += __shfl_xor(ps, 2);
      ps += __shfl_xor(ps, 4);
      ps += __shfl_xor(ps, 8);
      l[r] = l[r] * corr + ps;
      O[0][r] *= corr; O[1][r] *= corr; O[2][r] *= corr; O[3][r] *= corr;
      plds[w][r + 8 * hi][col]      = (__bf16)p0;
      plds[w][r + 8 * hi][16 + col] = (__bf16)p1;
    }

    // P as A-fragment (same-wave LDS store->load, in-order per ISA)
    v8bf c0 = *(const v8bf*)&plds[w][col][16 * hi];
    v8bf c1 = *(const v8bf*)&plds[w][col][16 * hi + 8];
    v16bf pa = __builtin_shufflevector(c0, c1, 0,1,2,3,4,5,6,7,8,9,10,11,12,13,14,15);

    O[0] = wmma_bf16(pa, vb0, O[0]);
    O[1] = wmma_bf16(pa, vb1, O[1]);
    O[2] = wmma_bf16(pa, vb2, O[2]);
    O[3] = wmma_bf16(pa, vb3, O[3]);
  };

  // prologue: buffer A <- kbase = 0
  v16bf kA0, kA1, kA2, kA3;
  {
    const __bf16* kr0 = Kh + (size_t)col * D_ + 16 * hi;
    const __bf16* kr1 = Kh + (size_t)(16 + col) * D_ + 16 * hi;
    kA0 = *(const v16bf*)(kr0);
    kA1 = *(const v16bf*)(kr0 + 32);
    kA2 = *(const v16bf*)(kr1);
    kA3 = *(const v16bf*)(kr1 + 32);
  }

  int kbase = 0;
  for (; kbase + 32 < kend; kbase += 64) {
    // prefetch buffer B <- kbase+32 (valid: kbase+32 < kend)
    v16bf kB0, kB1, kB2, kB3;
    {
      const __bf16* kr0 = Kh + (size_t)(kbase + 32 + col) * D_ + 16 * hi;
      const __bf16* kr1 = Kh + (size_t)(kbase + 48 + col) * D_ + 16 * hi;
      kB0 = *(const v16bf*)(kr0);
      kB1 = *(const v16bf*)(kr0 + 32);
      kB2 = *(const v16bf*)(kr1);
      kB3 = *(const v16bf*)(kr1 + 32);
    }
    step(kbase, kA0, kA1, kA2, kA3);

    // prefetch buffer A <- kbase+64 (clamped: always in-bounds)
    const int k2 = (kbase + 64 < kend) ? (kbase + 64) : 0;
    {
      const __bf16* kr0 = Kh + (size_t)(k2 + col) * D_ + 16 * hi;
      const __bf16* kr1 = Kh + (size_t)(k2 + 16 + col) * D_ + 16 * hi;
      kA0 = *(const v16bf*)(kr0);
      kA1 = *(const v16bf*)(kr0 + 32);
      kA2 = *(const v16bf*)(kr1);
      kA3 = *(const v16bf*)(kr1 + 32);
    }
    step(kbase + 32, kB0, kB1, kB2, kB3);
  }
  if (kbase < kend)
    step(kbase, kA0, kA1, kA2, kA3);

  // ---- normalize and write back to Y[B,T,C] (merging heads) ----
#pragma unroll
  for (int r = 0; r < 8; ++r) {
    const float inv = 1.0f / l[r];
    const int t = qbase + r + 8 * hi;
#pragma unroll
    for (int dt = 0; dt < 4; ++dt) {
      Y[((size_t)b * T_ + t) * C_ + h * D_ + dt * 16 + col] =
          (__bf16)(O[dt][r] * inv);
    }
  }
}

// ---------------------------------------------------------------------------
// Output projection GEMM: [B*T, C] x [C, C] (+bias) -> fp32 d_out.
// wT is W_proj^T [C][C].
// ---------------------------------------------------------------------------
__global__ __launch_bounds__(256) void k_proj_gemm(
    const __bf16* __restrict__ ybf, const __bf16* __restrict__ wT,
    const float* __restrict__ bias, float* __restrict__ out) {
  const int tiles_n = C_ / 64;                  // 12
  const int tm = blockIdx.x / tiles_n;
  const int tn = blockIdx.x % tiles_n;
  const int w    = threadIdx.x >> 5;
  const int lane = threadIdx.x & 31;
  const int col  = lane & 15;
  const int hi   = lane >> 4;
  const int mbase = tm * 256 + w * 32;
  const int nbase = tn * 64;

  const __bf16* arow0 = ybf + (size_t)(mbase + col) * C_;
  const __bf16* arow1 = ybf + (size_t)(mbase + 16 + col) * C_;
  const __bf16* brow[4];
#pragma unroll
  for (int nt = 0; nt < 4; ++nt)
    brow[nt] = wT + (size_t)(nbase + nt * 16 + col) * C_ + 16 * hi;

  v8f acc[8] = {};
  gemm_32x64_core(arow0, arow1, brow, hi, acc);

#pragma unroll
  for (int g = 0; g < 2; ++g) {
#pragma unroll
    for (int nt = 0; nt < 4; ++nt) {
      const int n  = nbase + nt * 16 + col;
      const float bv = bias[n];
#pragma unroll
      for (int r = 0; r < 8; ++r) {
        const int bt = mbase + g * 16 + r + 8 * hi;
        out[(size_t)bt * C_ + n] = acc[g * 4 + nt][r] + bv;
      }
    }
  }
}

// ---------------------------------------------------------------------------
extern "C" void kernel_launch(void* const* d_in, const int* in_sizes, int n_in,
                              void* d_out, int out_size, void* d_ws, size_t ws_size,
                              hipStream_t stream) {
  (void)in_sizes; (void)n_in; (void)out_size; (void)ws_size;
  const float* x      = (const float*)d_in[0];  // [B,T,C]
  const float* W_attn = (const float*)d_in[1];  // [C,3C]
  const float* b_attn = (const float*)d_in[2];  // [3C]
  const float* W_proj = (const float*)d_in[3];  // [C,C]
  const float* b_proj = (const float*)d_in[4];  // [C]
  float* out = (float*)d_out;                   // [B,T,C] fp32

  char* p = (char*)d_ws;
  __bf16* xbf    = (__bf16*)p; p += (size_t)B_ * T_ * C_ * 2;
  __bf16* wattnT = (__bf16*)p; p += (size_t)C3_ * C_ * 2;
  __bf16* wprojT = (__bf16*)p; p += (size_t)C_ * C_ * 2;
  __bf16* Qd     = (__bf16*)p; p += (size_t)B_ * T_ * C_ * 2;
  __bf16* Kd     = (__bf16*)p; p += (size_t)B_ * T_ * C_ * 2;
  __bf16* Vtd    = (__bf16*)p; p += (size_t)B_ * T_ * C_ * 2;
  __bf16* Ybf    = (__bf16*)p; p += (size_t)B_ * T_ * C_ * 2;

  const int nx = B_ * T_ * C_;
  k_f32_to_bf16<<<(nx + 255) / 256, 256, 0, stream>>>(x, xbf, nx);
  const int nwa = C_ * C3_;
  k_transpose_bf16<<<(nwa + 255) / 256, 256, 0, stream>>>(W_attn, wattnT, C_, C3_);
  const int nwp = C_ * C_;
  k_transpose_bf16<<<(nwp + 255) / 256, 256, 0, stream>>>(W_proj, wprojT, C_, C_);

  k_qkv_gemm<<<(B_ * T_ / 256) * (C3_ / 64), 256, 0, stream>>>(
      xbf, wattnT, b_attn, Qd, Kd, Vtd);

  k_attn<<<B_ * H_ * (T_ / 128), 256, 0, stream>>>(Qd, Kd, Vtd, Ybf);

  k_proj_gemm<<<(B_ * T_ / 256) * (C_ / 64), 256, 0, stream>>>(
      Ybf, wprojT, b_proj, out);
}
